// AlignedConv2d_90787018703586
// MI455X (gfx1250) — matrix operand order repaired and checked by reading, not compiled, not run on hardware
//
#include <hip/hip_runtime.h>
#include <cstdint>

typedef _Float16 half_t;
typedef __attribute__((ext_vector_type(16))) _Float16 v16h;
typedef __attribute__((ext_vector_type(8)))  _Float16 v8h;
typedef __attribute__((ext_vector_type(8)))  float    v8f;

constexpr int IMH = 320;
constexpr int IMW = 320;
constexpr int PLANE = IMH * IMW;

// ---------------- workspace layout (bytes) ----------------
constexpr size_t SZ_CH32 = (size_t)32 * PLANE * 2;            // one 32-ch f16 tensor
constexpr size_t OFF_T0   = 0;
constexpr size_t OFF_T1   = OFF_T0 + SZ_CH32;
constexpr size_t OFF_FEAT = OFF_T1 + SZ_CH32;                 // 64-ch f16 concat
constexpr size_t OFF_HID  = OFF_FEAT + 2 * SZ_CH32;
constexpr size_t OFF_Q16  = OFF_HID + SZ_CH32;
constexpr size_t OFF_R16  = OFF_Q16 + (size_t)3 * PLANE * 2;
constexpr size_t OFF_AFF  = OFF_R16 + (size_t)3 * PLANE * 2;  // 3-ch f32
constexpr size_t OFF_W1   = OFF_AFF + (size_t)3 * PLANE * 4;  // 32 x 96   f16
constexpr size_t OFF_WR11 = OFF_W1   + (size_t)32 * 96 * 2;   // 32 x 288
constexpr size_t OFF_WR12 = OFF_WR11 + (size_t)32 * 288 * 2;
constexpr size_t OFF_WP1  = OFF_WR12 + (size_t)32 * 288 * 2;  // 32 x 1600
constexpr size_t OFF_WPR1 = OFF_WP1  + (size_t)32 * 1600 * 2;
constexpr size_t OFF_WPR2 = OFF_WPR1 + (size_t)32 * 288 * 2;
// sampler prep tables alias FEAT (feat is dead after p1 conv):
constexpr size_t OFF_IDX  = OFF_FEAT;                         // [pos][4][PLANE] int
constexpr size_t OFF_WT   = OFF_FEAT + (size_t)16 * PLANE * 4;// [pos][4][PLANE] f32

__device__ __forceinline__ float lrelu_f(float v) { return v > 0.f ? v : 0.2f * v; }

// ---------------- fp32 -> fp16 cast ----------------
__global__ void k_cast_f16(const float* __restrict__ s, half_t* __restrict__ d, int n) {
  int i = blockIdx.x * 256 + threadIdx.x;
  if (i < n) d[i] = (half_t)s[i];
}

// ---------------- bicubic x2 (A = -0.75), matches reference _cubic ----------------
__device__ __forceinline__ float cubw(float t) {
  const float A = -0.75f;
  t = fabsf(t);
  float t2 = t * t;
  if (t <= 1.f) return ((A + 2.f) * t - (A + 3.f)) * t2 + 1.f;
  if (t < 2.f)  return A * (((t - 5.f) * t + 8.f) * t - 4.f);
  return 0.f;
}

__global__ void k_bicubic2x(const float* __restrict__ q, half_t* __restrict__ o) {
  int t = blockIdx.x * 256 + threadIdx.x;
  if (t >= 3 * PLANE) return;
  int c = t / PLANE;
  int r = t - c * PLANE;
  int y = r / IMW, x = r - y * IMW;
  float sy = (y + 0.5f) * 0.5f - 0.5f;
  float sx = (x + 0.5f) * 0.5f - 0.5f;
  int iy0 = (int)floorf(sy);
  int ix0 = (int)floorf(sx);
  const float* qc = q + (size_t)c * 160 * 160;
  float acc = 0.f;
#pragma unroll
  for (int a = 0; a < 4; ++a) {
    int ty = iy0 - 1 + a;
    float wy = cubw(sy - (float)ty);
    int tyc = ty < 0 ? 0 : (ty > 159 ? 159 : ty);
#pragma unroll
    for (int b = 0; b < 4; ++b) {
      int tx = ix0 - 1 + b;
      float wx = cubw(sx - (float)tx);
      int txc = tx < 0 ? 0 : (tx > 159 ? 159 : tx);
      acc += wy * wx * qc[tyc * 160 + txc];
    }
  }
  o[t] = (half_t)acc;
}

// ---------------- weight repack: (32, K) f32 -> (32, KPAD) f16, zero padded ----------------
__global__ void k_repack_w(const float* __restrict__ w, half_t* __restrict__ wp, int K, int KPAD) {
  int i = blockIdx.x * 256 + threadIdx.x;
  if (i >= 32 * KPAD) return;
  int co = i / KPAD;
  int k  = i - co * KPAD;
  wp[i] = (k < K) ? (half_t)w[co * K + k] : (half_t)0.f;
}

// ---------------- implicit-GEMM conv via V_WMMA_F32_16X16X32_F16 ----------------
// WG = 128 threads (4 wave32s): 32 output channels x 64 pixels (one row).
// Double-buffered LDS; A (weight) tile staged with GLOBAL_LOAD_ASYNC_TO_LDS_B128
// (ASYNCcnt), B (im2col) tile staged/transposed with scalar stores; padded LDS
// strides (40 halfs) make the fragment ds_load_b128s bank-conflict-free.
// ACT: 0 = lrelu, 1 = relu, 2 = (+bias, +residual, lrelu)
template<int CIN, int KH, int KW, int PAD, int ACT>
__global__ __launch_bounds__(128) void k_conv_wmma(
    const half_t* __restrict__ Xin, const half_t* __restrict__ Wp,
    const float* __restrict__ bias, const half_t* __restrict__ Res,
    half_t* __restrict__ Yout)
{
  constexpr int K = CIN * KH * KW;
  constexpr int KPAD = (K + 31) & ~31;
  constexpr int KSTEPS = KPAD / 32;
  constexpr int AST = 40;   // LDS row stride (halfs), padded vs 32
  constexpr int BST = 40;

  __shared__ alignas(16) half_t ldsA[2][32 * AST];
  __shared__ alignas(16) half_t ldsB[2][64 * BST];

  const int tid  = threadIdx.x;
  const int lane = tid & 31;
  const int wave = tid >> 5;
  const int y0 = blockIdx.y;
  const int x0 = blockIdx.x * 64;

  const int m    = wave >> 1;          // 0..1 : co tile
  const int ncol = (wave & 1) * 2;     // base n-tile (0 or 2)
  const int hsel = lane >> 4;          // half-wave select
  const int l16  = lane & 15;

  const int a_co  = tid >> 2;          // 0..31
  const int a_seg = (tid & 3) * 8;     // 0,8,16,24
  const int b_kk  = tid >> 2;          // 0..31  (k within step)
  const int b_q   = tid & 3;           // n interleave base

  v8f acc[2] = {};

  auto stage = [&](int ks, int buf) {
    const int k0 = ks * 32;
    // ---- A tile: one 16B async global->LDS DMA per thread
    {
      unsigned la = (unsigned)(uintptr_t)&ldsA[buf][a_co * AST + a_seg];
      unsigned long long ga =
          (unsigned long long)(uintptr_t)(Wp + (size_t)a_co * KPAD + k0 + a_seg);
      asm volatile("global_load_async_to_lds_b128 %0, %1, off"
                   :: "v"(la), "v"(ga) : "memory");
    }
    // ---- B tile: im2col with transpose into [n][k] rows
    {
      int kg  = k0 + b_kk;
      int ci  = kg / (KH * KW);
      int rem = kg - ci * (KH * KW);
      int ky  = rem / KW;
      int kx  = rem - ky * KW;
      int yy  = y0 + ky - PAD;
      bool rowok = (ci < CIN) && ((unsigned)yy < (unsigned)IMH);
      const half_t* src = Xin + (size_t)ci * PLANE + (size_t)yy * IMW;
      int xb = x0 + kx - PAD;
#pragma unroll
      for (int i = 0; i < 16; ++i) {
        int n  = b_q + 4 * i;          // interleaved n -> 16 distinct store banks
        int xx = xb + n;
        half_t v = (half_t)0.f;
        if (rowok && (unsigned)xx < (unsigned)IMW) v = src[xx];
        ldsB[buf][n * BST + b_kk] = v;
      }
      // L2 warm-up for the row needed two K-steps from now (global_prefetch_b8)
      int kg2 = kg + 64;
      if (kg2 < KPAD) {
        int ci2  = kg2 / (KH * KW);
        int rem2 = kg2 - ci2 * (KH * KW);
        int ky2  = rem2 / KW;
        int kx2  = rem2 - ky2 * KW;
        int yy2  = y0 + ky2 - PAD;
        if ((ci2 < CIN) && (unsigned)yy2 < (unsigned)IMH) {
          int xp = x0 + kx2 - PAD;
          xp = xp < 0 ? 0 : xp;
          __builtin_prefetch(Xin + (size_t)ci2 * PLANE + (size_t)yy2 * IMW + xp, 0, 3);
        }
      }
    }
  };

  stage(0, 0);

  for (int ks = 0; ks < KSTEPS; ++ks) {
    const int buf = ks & 1;
    // complete our async DMA writes for buffer `buf`, then publish to the WG
    asm volatile("s_wait_asynccnt 0x0" ::: "memory");
    __syncthreads();
    // overlap: stage the next K-step into the other buffer while doing WMMAs
    if (ks + 1 < KSTEPS) stage(ks + 1, buf ^ 1);

    // A fragment (16x32 f16): lanes 0-15 hold K 0-7 & 16-23 of row M=l16,
    // lanes 16-31 hold K 8-15 & 24-31.  Two ds_load_b128 per lane.
    const half_t* rowA = &ldsA[buf][(m * 16 + l16) * AST];
    v8h alo = *(const v8h*)(rowA + hsel * 8);
    v8h ahi = *(const v8h*)(rowA + 16 + hsel * 8);
    v16h afrag = __builtin_shufflevector(alo, ahi,
        0,1,2,3,4,5,6,7,8,9,10,11,12,13,14,15);

#pragma unroll
    for (int t = 0; t < 2; ++t) {
      // B fragment (32x16): lanes 0-15 hold K 0-7 & 16-23 of col N=l16, lanes 16-31 K 8-15 & 24-31.
      const half_t* rowB = &ldsB[buf][((ncol + t) * 16 + l16) * BST + hsel * 8];
      v8h blo = *(const v8h*)(rowB);
      v8h bhi = *(const v8h*)(rowB + 16);
      v16h bfrag = __builtin_shufflevector(blo, bhi,
          0,1,2,3,4,5,6,7,8,9,10,11,12,13,14,15);
      acc[t] = __builtin_amdgcn_wmma_f32_16x16x32_f16(
          false, afrag, false, bfrag, (short)0, acc[t], false, false);
    }
  }

  // epilogue: D layout -> lane<16: M=i, lane>=16: M=8+i ; N = l16
#pragma unroll
  for (int t = 0; t < 2; ++t) {
    int xg = x0 + (ncol + t) * 16 + l16;
#pragma unroll
    for (int i = 0; i < 8; ++i) {
      int co = m * 16 + hsel * 8 + i;
      float v = acc[t][i] + bias[co];
      if (ACT == 2) v += (float)Res[(size_t)co * PLANE + (size_t)y0 * IMW + xg];
      if (ACT == 1) v = v > 0.f ? v : 0.f;
      else          v = lrelu_f(v);
      Yout[(size_t)co * PLANE + (size_t)y0 * IMW + xg] = (half_t)v;
    }
  }
}

// ---------------- 1x1 conv -> affine params (clipped) ----------------
__global__ void k_affine(const half_t* __restrict__ hid, const float* __restrict__ w,
                         const float* __restrict__ b, float* __restrict__ aff) {
  int p = blockIdx.x * 256 + threadIdx.x;
  if (p >= PLANE) return;
  float s0 = 0.f, s1 = 0.f, s2 = 0.f;
#pragma unroll
  for (int ci = 0; ci < 32; ++ci) {
    float h = (float)hid[(size_t)ci * PLANE + p];
    s0 += w[ci] * h;
    s1 += w[32 + ci] * h;
    s2 += w[64 + ci] * h;
  }
  s0 = fminf(fmaxf(s0 + b[0] + 1.f, -3.f), 3.f);
  s1 = fminf(fmaxf(s1 + b[1] + 1.f, -3.f), 3.f);
  s2 = fminf(fmaxf(s2 + b[2] + 1.f, -3.f), 3.f);
  aff[p] = s0;
  aff[PLANE + p] = s1;
  aff[2 * PLANE + p] = s2;
}

// ---------------- sampler prep: per (pos, pixel) 4 gather indices + 4 weights ----------------
__device__ __forceinline__ int reflect320(int a) {
  a = a < 0 ? -a : a;
  return a >= IMH ? 2 * IMH - 2 - a : a;   // pad-1 reflect of 320
}

__global__ void k_prep(const float* __restrict__ aff, int* __restrict__ IDX,
                       float* __restrict__ WT) {
  int t = blockIdx.x * 256 + threadIdx.x;
  if (t >= 4 * PLANE) return;
  int pos = t / PLANE;
  int p = t - pos * PLANE;
  int i = p / IMW, j = p - i * IMW;
  float sx = aff[p], sy = aff[PLANE + p];
  float theta = (aff[2 * PLANE + p] - 1.f) * 1.0472f;
  float px = ((pos & 2) ? 0.5f : -0.5f) * sx;   // pn_x = [-.5,-.5,.5,.5]
  float py = ((pos & 1) ? 0.5f : -0.5f) * sy;   // pn_y = [-.5,.5,-.5,.5]
  float ct = cosf(theta), st = sinf(theta);
  float rx = px * ct - py * st;
  float ry = px * st + py * ct;
  float p_x = rx + 0.5f + (float)(i + 1);
  float p_y = ry + 0.5f + (float)(j + 1);
  float ltx = floorf(p_x), lty = floorf(p_y);
  float rbx = ltx + 1.f,  rby = lty + 1.f;
  const float MX = 321.f;                       // hp-1 = wp-1 = 321
  float ltxc = fminf(fmaxf(ltx, 0.f), MX);
  float ltyc = fminf(fmaxf(lty, 0.f), MX);
  float rbxc = fminf(fmaxf(rbx, 0.f), MX);
  float rbyc = fminf(fmaxf(rby, 0.f), MX);
  p_x = fminf(fmaxf(p_x, 0.f), MX);
  p_y = fminf(fmaxf(p_y, 0.f), MX);
  float g_lt = (1.f + ltxc - p_x) * (1.f + ltyc - p_y);
  float g_rb = (1.f - (rbxc - p_x)) * (1.f - (rbyc - p_y));
  float g_lb = (1.f + ltxc - p_x) * (1.f - (rbyc - p_y));
  float g_rt = (1.f - (rbxc - p_x)) * (1.f + ltyc - p_y);
  int ax = reflect320((int)ltxc - 1), ay = reflect320((int)ltyc - 1);
  int bx = reflect320((int)rbxc - 1), by = reflect320((int)rbyc - 1);
  size_t base = (size_t)(pos * 4) * PLANE + p;
  IDX[base]                      = ax * IMW + ay;
  IDX[base + (size_t)PLANE]      = bx * IMW + by;
  IDX[base + (size_t)2 * PLANE]  = ax * IMW + by;
  IDX[base + (size_t)3 * PLANE]  = bx * IMW + ay;
  WT[base]                       = g_lt;
  WT[base + (size_t)PLANE]       = g_rb;
  WT[base + (size_t)2 * PLANE]   = g_lb;
  WT[base + (size_t)3 * PLANE]   = g_rt;
}

// ---------------- gather + float2-coalesced scatter to (64,640,640) ----------------
__global__ void k_sample(const float* __restrict__ X, const int* __restrict__ IDX,
                         const float* __restrict__ WT, float* __restrict__ out) {
  int p = blockIdx.x * 256 + threadIdx.x;
  if (p >= PLANE) return;
  int u = blockIdx.y;     // output row parity
  int c = blockIdx.z;     // channel
  int i = p / IMW, j = p - i * IMW;
  const float* xc = X + (size_t)c * PLANE;
  float2 r;
#pragma unroll
  for (int v = 0; v < 2; ++v) {
    int pos = 2 * u + v;
    size_t base = (size_t)(pos * 4) * PLANE + p;
    float acc = 0.f;
#pragma unroll
    for (int k = 0; k < 4; ++k)
      acc += WT[base + (size_t)k * PLANE] * xc[IDX[base + (size_t)k * PLANE]];
    if (v == 0) r.x = acc; else r.y = acc;
  }
  *(float2*)(out + (size_t)c * 640 * 640 + (size_t)(2 * i + u) * 640 + 2 * j) = r;
}

extern "C" void kernel_launch(void* const* d_in, const int* in_sizes, int n_in,
                              void* d_out, int out_size, void* d_ws, size_t ws_size,
                              hipStream_t stream) {
  (void)in_sizes; (void)n_in; (void)out_size; (void)ws_size;
  const float* x       = (const float*)d_in[0];
  const float* query   = (const float*)d_in[1];
  const float* ref     = (const float*)d_in[2];
  const float* conv1_w = (const float*)d_in[3];
  const float* conv1_b = (const float*)d_in[4];
  const float* r1_w1   = (const float*)d_in[5];
  const float* r1_b1   = (const float*)d_in[6];
  const float* r1_w2   = (const float*)d_in[7];
  const float* r1_b2   = (const float*)d_in[8];
  const float* p1_w    = (const float*)d_in[9];
  const float* p1_b    = (const float*)d_in[10];
  const float* pr_w1   = (const float*)d_in[11];
  const float* pr_b1   = (const float*)d_in[12];
  const float* pr_w2   = (const float*)d_in[13];
  const float* pr_b2   = (const float*)d_in[14];
  const float* p2_w    = (const float*)d_in[15];
  const float* p2_b    = (const float*)d_in[16];
  float* out = (float*)d_out;

  char* ws = (char*)d_ws;
  half_t* T0    = (half_t*)(ws + OFF_T0);
  half_t* T1    = (half_t*)(ws + OFF_T1);
  half_t* FEAT  = (half_t*)(ws + OFF_FEAT);
  half_t* HID   = (half_t*)(ws + OFF_HID);
  half_t* Q16   = (half_t*)(ws + OFF_Q16);
  half_t* R16   = (half_t*)(ws + OFF_R16);
  float*  AFF   = (float*)(ws + OFF_AFF);
  half_t* W1p   = (half_t*)(ws + OFF_W1);
  half_t* WR11p = (half_t*)(ws + OFF_WR11);
  half_t* WR12p = (half_t*)(ws + OFF_WR12);
  half_t* WP1p  = (half_t*)(ws + OFF_WP1);
  half_t* WPR1p = (half_t*)(ws + OFF_WPR1);
  half_t* WPR2p = (half_t*)(ws + OFF_WPR2);
  int*    IDX   = (int*)(ws + OFF_IDX);
  float*  WT    = (float*)(ws + OFF_WT);

  dim3 blk(256);
  k_cast_f16 <<<(3 * PLANE + 255) / 256, blk, 0, stream>>>(ref, R16, 3 * PLANE);
  k_bicubic2x<<<(3 * PLANE + 255) / 256, blk, 0, stream>>>(query, Q16);

  k_repack_w<<<(32 * 96   + 255) / 256, blk, 0, stream>>>(conv1_w, W1p,   75,   96);
  k_repack_w<<<(32 * 288  + 255) / 256, blk, 0, stream>>>(r1_w1,   WR11p, 288,  288);
  k_repack_w<<<(32 * 288  + 255) / 256, blk, 0, stream>>>(r1_w2,   WR12p, 288,  288);
  k_repack_w<<<(32 * 1600 + 255) / 256, blk, 0, stream>>>(p1_w,    WP1p,  1600, 1600);
  k_repack_w<<<(32 * 288  + 255) / 256, blk, 0, stream>>>(pr_w1,   WPR1p, 288,  288);
  k_repack_w<<<(32 * 288  + 255) / 256, blk, 0, stream>>>(pr_w2,   WPR2p, 288,  288);

  dim3 cgrid(IMW / 64, IMH);
  dim3 cblk(128);
  // query head -> FEAT channels 32..63
  hipLaunchKernelGGL(HIP_KERNEL_NAME(k_conv_wmma<3,5,5,2,0>),  cgrid, cblk, 0, stream,
                     Q16, W1p, conv1_b, (const half_t*)nullptr, T0);
  hipLaunchKernelGGL(HIP_KERNEL_NAME(k_conv_wmma<32,3,3,1,1>), cgrid, cblk, 0, stream,
                     T0, WR11p, r1_b1, (const half_t*)nullptr, T1);
  hipLaunchKernelGGL(HIP_KERNEL_NAME(k_conv_wmma<32,3,3,1,2>), cgrid, cblk, 0, stream,
                     T1, WR12p, r1_b2, (const half_t*)T0, FEAT + (size_t)32 * PLANE);
  // ref head -> FEAT channels 0..31
  hipLaunchKernelGGL(HIP_KERNEL_NAME(k_conv_wmma<3,5,5,2,0>),  cgrid, cblk, 0, stream,
                     R16, W1p, conv1_b, (const half_t*)nullptr, T0);
  hipLaunchKernelGGL(HIP_KERNEL_NAME(k_conv_wmma<32,3,3,1,1>), cgrid, cblk, 0, stream,
                     T0, WR11p, r1_b1, (const half_t*)nullptr, T1);
  hipLaunchKernelGGL(HIP_KERNEL_NAME(k_conv_wmma<32,3,3,1,2>), cgrid, cblk, 0, stream,
                     T1, WR12p, r1_b2, (const half_t*)T0, FEAT);
  // prediction branch
  hipLaunchKernelGGL(HIP_KERNEL_NAME(k_conv_wmma<64,5,5,2,0>), cgrid, cblk, 0, stream,
                     FEAT, WP1p, p1_b, (const half_t*)nullptr, T0);
  hipLaunchKernelGGL(HIP_KERNEL_NAME(k_conv_wmma<32,3,3,1,1>), cgrid, cblk, 0, stream,
                     T0, WPR1p, pr_b1, (const half_t*)nullptr, T1);
  hipLaunchKernelGGL(HIP_KERNEL_NAME(k_conv_wmma<32,3,3,1,2>), cgrid, cblk, 0, stream,
                     T1, WPR2p, pr_b2, (const half_t*)T0, HID);

  k_affine<<<(PLANE + 255) / 256, blk, 0, stream>>>(HID, p2_w, p2_b, AFF);
  k_prep  <<<(4 * PLANE + 255) / 256, blk, 0, stream>>>(AFF, IDX, WT);

  dim3 sgrid((PLANE + 255) / 256, 2, 64);
  k_sample<<<sgrid, blk, 0, stream>>>(x, IDX, WT, out);
}
